// IFA_MultiLevel_32160715113220
// MI455X (gfx1250) — compile-verified, matrix operand order checked by reading, not run on hardware
//
#include <hip/hip_runtime.h>

typedef __attribute__((ext_vector_type(16))) __bf16 v16bf;
typedef __attribute__((ext_vector_type(8)))  float  v8f;
typedef __attribute__((ext_vector_type(4)))  unsigned int v4u;
typedef __attribute__((ext_vector_type(8)))  int v8i;
typedef __attribute__((ext_vector_type(4)))  int v4i;
typedef __bf16 bf16_t;

#define BSZ    4
#define CIN    256
#define HDIM   128
#define NPIX   (BSZ * HDIM * HDIM)   // 65536 output pixels
#define OUTCH  256
#define KSTEPS 33                    // 32 feature K-steps (4 levels x 8) + 1 extras step
#define MTILE  64                    // pixels per workgroup

// ---------------------------------------------------------------------------
// Pack W (256 x 1040 f32, ref layout: per-level [rel(2), feat(256), scale(2)])
// into bf16 Wb[s][n][32], s = 0..32.  K' layout: k'<1024 -> level feats
// (level = k'>>8, cin = k'&255); k' in [1024,1040) -> extras
// [l0:rel_y,rel_x,scale_y,scale_x, l1:..., ...]; k' >= 1040 -> zero pad.
// ---------------------------------------------------------------------------
__global__ void ifa_pack_w_kernel(const float* __restrict__ W,
                                  bf16_t* __restrict__ Wb) {
  int idx = blockIdx.x * 256 + threadIdx.x;   // 33*256*32 = 270336 elements
  int kk = idx & 31;
  int n  = (idx >> 5) & 255;
  int s  = idx >> 13;
  int kp = s * 32 + kk;
  float v = 0.0f;
  if (kp < 1024) {
    int l = kp >> 8, c = kp & 255;
    v = W[n * 1040 + l * 260 + 2 + c];
  } else if (kp < 1040) {
    int e = kp - 1024, l = e >> 2, j = e & 3;
    // j=0: rel_y (k=l*260+0), j=1: rel_x (+1), j=2: scale_y (+258), j=3: scale_x (+259)
    v = W[n * 1040 + l * 260 + ((j < 2) ? j : (256 + j))];
  }
  Wb[idx] = (bf16_t)v;
}

// ---------------------------------------------------------------------------
// Fused gather + bf16 WMMA GEMM + bias/BN/ReLU epilogue.
// Grid: NPIX/MTILE = 1024 workgroups, 256 threads (8 wave32).
// WG tile: 256 out-channels (M) x 64 pixels (N), double-buffered LDS.
// Weight blocks (16KB bf16) staged by the Tensor Data Mover (TENSORcnt);
// feature blocks gathered+converted by the wave (DScnt).
// ---------------------------------------------------------------------------
__global__ __launch_bounds__(256) void ifa_wmma_gemm_kernel(
    const float* __restrict__ x0, const float* __restrict__ x1,
    const float* __restrict__ x2, const float* __restrict__ x3,
    const bf16_t* __restrict__ Wb,
    const float* __restrict__ bias, const float* __restrict__ bn_gamma,
    const float* __restrict__ bn_beta, const float* __restrict__ bn_mean,
    const float* __restrict__ bn_var, float* __restrict__ out) {

  // Padded rows: 40 halves = 80B (16B-aligned, bank-conflict-free strides).
  // TDM produces the padding itself via pad_interval/pad_amount.
  __shared__ bf16_t As[2][OUTCH][40];   // weight blocks: 256 ch x 32 k
  __shared__ bf16_t Bsh[2][MTILE][40];  // feature blocks: 64 px x 32 k
  __shared__ float alphaS[OUTCH];
  __shared__ float shiftS[OUTCH];

  const int tid    = threadIdx.x;
  const int lane   = tid & 31;
  const int wave   = tid >> 5;
  const int m_base = blockIdx.x * MTILE;

  // Fold conv-bias + eval-mode BN into per-channel scale/shift.
  {
    float inv = bn_gamma[tid] * rsqrtf(bn_var[tid] + 1e-5f);
    alphaS[tid] = inv;
    shiftS[tid] = (bias[tid] - bn_mean[tid]) * inv + bn_beta[tid];
  }

  // Per-thread pixel assignment for feature staging.
  const int p    = tid & 63;         // pixel within tile
  const int kg   = tid >> 6;         // 0..3: which 8-channel group
  const int mloc = m_base + p;
  const int bb   = mloc >> 14;
  const int yy   = (mloc >> 7) & 127;
  const int xx   = mloc & 127;

  const int koff = (lane & 16) ? 8 : 0;   // K-half per lane group (ISA layout)
  const int lcol = lane & 15;

  // ---- TDM: DMA one 256x32 bf16 weight block into As[buf] with 16B row pad.
  auto tdm_load_w = [&](int s, int buf) {
    unsigned long long ga =
        (unsigned long long)(uintptr_t)(Wb + ((size_t)s << 13));
    unsigned lds_off =
        (unsigned)(unsigned long long)(uintptr_t)(&As[buf][0][0]);
    v4u g0;
    g0[0] = 1u;                                   // count=1, no gather
    g0[1] = lds_off;                              // lds_addr
    g0[2] = (unsigned)ga;                         // global_addr[31:0]
    g0[3] = ((unsigned)(ga >> 32) & 0x01FFFFFFu)  // global_addr[56:32]
            | (2u << 30);                         // type=2 ("image")
    v8i g1;
    g1[0] = (1 << 16)      // data_size = 2 bytes
          | (1 << 20)      // pad_enable
          | (3 << 22)      // pad_interval: every 16 DWORDs (64B)
          | (3 << 25);     // pad_amount: 4 DWORDs (16B)
    g1[1] = 32 << 16;      // tensor_dim0 = 32 (bits 63:48)
    g1[2] = 256 << 16;     // tensor_dim1 = 256 (bits 95:80)
    g1[3] = 32 << 16;      // tile_dim0 = 32 (bits 127:112)
    g1[4] = 256;           // tile_dim1 = 256 (bits 143:128), tile_dim2 = 0
    g1[5] = 32;            // tensor_dim0_stride = 32
    g1[6] = 0;
    g1[7] = 0;
    v4i z4 = {0, 0, 0, 0};
    v8i z8 = {0, 0, 0, 0, 0, 0, 0, 0};
    __builtin_amdgcn_tensor_load_to_lds(g0, g1, z4, z4, z8, 0);
  };

  // ---- feature gather (+analytic rel/scale extras) -> Bsh[buf]
  auto stage_feats = [&](int s, int buf) {
    if (s < 32) {
      const int l  = s >> 3;                 // pyramid level
      const int hs = HDIM >> l;              // level size (128,64,32,16)
      const float* xl = (l == 0) ? x0 : (l == 1) ? x1 : (l == 2) ? x2 : x3;
      // nearest grid-sample == integer downshift for these sizes
      const float* src = xl + (((size_t)bb * CIN * hs + (size_t)(yy >> l)) * hs
                               + (size_t)(xx >> l));
      const int cbase = ((s & 7) << 5) + (kg << 3);
      const size_t cstride = (size_t)hs * hs;
#pragma unroll
      for (int j = 0; j < 8; ++j)
        Bsh[buf][p][kg * 8 + j] = (bf16_t)src[(size_t)(cbase + j) * cstride];
    } else {
      // extras step: [l: rel_y, rel_x, scale_y, scale_x] x 4 levels, then zeros
#pragma unroll
      for (int j = 0; j < 8; ++j) {
        int k = kg * 8 + j;
        float v = 0.0f;
        if (k < 16) {
          int l = k >> 2, jj = k & 3;
          float hs = (float)(HDIM >> l);
          if (jj == 0)
            v = (float)(2 * yy + 1) * hs * (1.0f / HDIM) - (float)(2 * (yy >> l) + 1);
          else if (jj == 1)
            v = (float)(2 * xx + 1) * hs * (1.0f / HDIM) - (float)(2 * (xx >> l) + 1);
          else
            v = (2.0f / 256.0f) * (float)HDIM;   // scale_y == scale_x == 1.0
        }
        Bsh[buf][p][k] = (bf16_t)v;
      }
    }
  };

  v8f acc[2][4];
#pragma unroll
  for (int i = 0; i < 2; ++i)
#pragma unroll
    for (int j = 0; j < 4; ++j)
#pragma unroll
      for (int r = 0; r < 8; ++r) acc[i][j][r] = 0.0f;

  // ---- prologue: stage step 0 into buffer 0
  if (wave == 0) tdm_load_w(0, 0);
  stage_feats(0, 0);
  if (wave == 0) __builtin_amdgcn_s_wait_tensorcnt(0);
  __syncthreads();

  for (int s = 0; s < KSTEPS; ++s) {
    const int cur = s & 1;
    const int nxt = cur ^ 1;

    // ---- kick off next block's staging before computing (overlap with WMMA)
    if (s + 1 < KSTEPS) {
      if (wave == 0) tdm_load_w(s + 1, nxt);
      stage_feats(s + 1, nxt);
    }

    // ---- build fragments per 16-bit WMMA VGPR layout and multiply
    union V16 { v16bf v; uint4 q[2]; };
    V16 af[2], bfr[4];
#pragma unroll
    for (int i = 0; i < 2; ++i) {
      const bf16_t* ap = &As[cur][(2 * wave + i) * 16 + lcol][koff];
      af[i].q[0] = *(const uint4*)ap;          // K {0..7} / {8..15}
      af[i].q[1] = *(const uint4*)(ap + 16);   // K {16..23} / {24..31}
    }
#pragma unroll
    for (int j = 0; j < 4; ++j) {
      const bf16_t* bp = &Bsh[cur][j * 16 + lcol][koff];
      bfr[j].q[0] = *(const uint4*)bp;
      bfr[j].q[1] = *(const uint4*)(bp + 16);
    }
#pragma unroll
    for (int i = 0; i < 2; ++i)
#pragma unroll
      for (int j = 0; j < 4; ++j)
        acc[i][j] = __builtin_amdgcn_wmma_f32_16x16x32_bf16(
            false, af[i].v, false, bfr[j].v, (short)0, acc[i][j], false, false);

    // ---- publish next buffer: TDM done (wave0), ds_stores done (barrier)
    if (s + 1 < KSTEPS && wave == 0) __builtin_amdgcn_s_wait_tensorcnt(0);
    __syncthreads();
  }

  // ---- epilogue: scale/shift/ReLU; stores coalesced over pixels (lanes 0..15)
  const int lhalf = lane >> 4;  // C/D layout: lanes 16..31 hold rows M+8
#pragma unroll
  for (int i = 0; i < 2; ++i) {
#pragma unroll
    for (int j = 0; j < 4; ++j) {
      int pix  = m_base + j * 16 + lcol;
      int ob   = pix >> 14;
      int prem = pix & 16383;
      int chb  = (2 * wave + i) * 16 + 8 * lhalf;
#pragma unroll
      for (int r = 0; r < 8; ++r) {
        int ch  = chb + r;
        float v = acc[i][j][r] * alphaS[ch] + shiftS[ch];
        out[((size_t)ob * OUTCH + ch) * (size_t)(HDIM * HDIM) + prem] = fmaxf(v, 0.0f);
      }
    }
  }
}

// ---------------------------------------------------------------------------
extern "C" void kernel_launch(void* const* d_in, const int* in_sizes, int n_in,
                              void* d_out, int out_size, void* d_ws, size_t ws_size,
                              hipStream_t stream) {
  (void)in_sizes; (void)n_in; (void)out_size; (void)ws_size;
  const float* x0 = (const float*)d_in[0];
  const float* x1 = (const float*)d_in[1];
  const float* x2 = (const float*)d_in[2];
  const float* x3 = (const float*)d_in[3];
  const float* W  = (const float*)d_in[4];
  const float* b  = (const float*)d_in[5];
  const float* g  = (const float*)d_in[6];
  const float* be = (const float*)d_in[7];
  const float* mn = (const float*)d_in[8];
  const float* vr = (const float*)d_in[9];
  float* out = (float*)d_out;

  bf16_t* Wb = (bf16_t*)d_ws;   // 33*256*32*2 = 540,672 bytes

  ifa_pack_w_kernel<<<dim3((KSTEPS * 256 * 32) / 256), dim3(256), 0, stream>>>(W, Wb);
  ifa_wmma_gemm_kernel<<<dim3(NPIX / MTILE), dim3(256), 0, stream>>>(
      x0, x1, x2, x3, Wb, b, g, be, mn, vr, out);
}
